// RNNStack_6579889897990
// MI455X (gfx1250) — matrix operand order repaired and checked by reading, not compiled, not run on hardware
//
#include <hip/hip_runtime.h>
#include <hip/hip_bf16.h>
#include <stdint.h>

// Problem constants (match reference)
#define B_  32
#define T_  1024
#define H_  512
#define G_  1536   // 3*H
#define L_  3
#define LN_EPS 1e-5f

typedef __attribute__((ext_vector_type(16))) __bf16 v16bf;
typedef __attribute__((ext_vector_type(8)))  float  v8f;

// gfx1250 async global->LDS copy path (ASYNCcnt), with portable fallback.
#if defined(__gfx1250__) && \
    __has_builtin(__builtin_amdgcn_global_load_async_to_lds_b128) && \
    __has_builtin(__builtin_amdgcn_s_wait_asynccnt)
#define USE_ASYNC_LDS 1
#define GLOBAL_AS __attribute__((address_space(1)))
#define LDS_AS    __attribute__((address_space(3)))
typedef int v4i_async __attribute__((vector_size(16)));
typedef GLOBAL_AS v4i_async* gv4i_p;   // global int4*
typedef LDS_AS    v4i_async* lv4i_p;   // LDS int4*
#endif

// ---------------------------------------------------------------------------
// fp32 -> bf16 cast (grid-stride)
// ---------------------------------------------------------------------------
__global__ void cast_f32_to_bf16_kernel(const float* __restrict__ src,
                                        __bf16* __restrict__ dst, int n) {
  int i = blockIdx.x * blockDim.x + threadIdx.x;
  int stride = gridDim.x * blockDim.x;
  for (; i < n; i += stride) dst[i] = (__bf16)src[i];
}

// ---------------------------------------------------------------------------
// Input projection: xp[m, g] = sum_k xin[m,k]*Wih[g,k] + bih[g]
// M = B*T = 32768, N = G = 1536, K = H = 512.  One 16x16 tile per wave,
// K swept in steps of 32 with v_wmma_f32_16x16x32_bf16.
// Fragment layouts follow CDNA5 ISA 7.12.2 exactly.
// ---------------------------------------------------------------------------
__global__ __launch_bounds__(256) void xproj_kernel(
    const __bf16* __restrict__ xin,   // [B*T, H] row-major
    const __bf16* __restrict__ wih,   // [G, H] row-major  (== B matrix [K,N] col-major)
    const float*  __restrict__ bih,   // [G]
    float* __restrict__ xp)           // [B*T, G]
{
  const int lane = threadIdx.x & 31;
  const int wave = threadIdx.x >> 5;
  const int wid  = blockIdx.x * 8 + wave;
  const int NT   = G_ / 16;            // 96 N-tiles
  const int mt   = wid / NT;
  const int nt   = wid % NT;
  const int m0   = mt * 16;
  const int n0   = nt * 16;

  // A-matrix 16x32 bf16: lane<16 -> M=lane, K chunks [0..7] & [16..23];
  //                      lane>=16 -> same M, K chunks [8..15] & [24..31]
  const int arow = m0 + (lane & 15);
  const int akb  = (lane >> 4) * 8;
  // B-matrix 32x16 bf16: lane&15 -> N column, lane>>4 selects K chunks [0..7]&[16..23] or [8..15]&[24..31]
  const int bn   = n0 + (lane & 15);
  const int bkb  = (lane >> 4) * 8;

  v8f acc = {};
  #pragma unroll
  for (int k0 = 0; k0 < H_; k0 += 32) {
    union { v16bf v; uint4 q[2]; } a, b;
    const __bf16* ap = xin + (size_t)arow * H_ + k0 + akb;
    a.q[0] = *(const uint4*)(ap);
    a.q[1] = *(const uint4*)(ap + 16);
    const __bf16* bp = wih + (size_t)bn * H_ + k0 + bkb;
    b.q[0] = *(const uint4*)(bp);
    b.q[1] = *(const uint4*)(bp + 16);
    acc = __builtin_amdgcn_wmma_f32_16x16x32_bf16(false, a.v, false, b.v,
                                                  (short)0, acc, false, false);
  }
  // D layout: VGPR r -> M = r (lanes 0-15) / 8+r (lanes 16-31); N = lane&15
  const int   n     = n0 + (lane & 15);
  const float bias  = bih[n];
  const int   mbase = m0 + (lane >> 4) * 8;
  #pragma unroll
  for (int r = 0; r < 8; ++r)
    xp[(size_t)(mbase + r) * G_ + n] = acc[r] + bias;
}

// ---------------------------------------------------------------------------
// Per-layer state init: h32/hbf <- init_state, reset grid barrier words
// ---------------------------------------------------------------------------
__global__ void init_layer_kernel(const float* __restrict__ h0,
                                  float* __restrict__ h32,
                                  __bf16* __restrict__ hbf,
                                  unsigned* __restrict__ bar) {
  int i = blockIdx.x * blockDim.x + threadIdx.x;
  if (i == 0) { bar[0] = 0u; bar[1] = 0u; }
  if (i < B_ * H_) { float v = h0[i]; h32[i] = v; hbf[i] = (__bf16)v; }
}

// ---------------------------------------------------------------------------
// Helper: async (or fallback) copy of n16 16-byte chunks global -> LDS,
// distributed over the block.  Caller must s_wait_asynccnt + __syncthreads.
// ---------------------------------------------------------------------------
__device__ __forceinline__ void stage_to_lds(const __bf16* __restrict__ gsrc,
                                             __bf16* __restrict__ ldst,
                                             int n16, int tid, int nthreads) {
#ifdef USE_ASYNC_LDS
  for (int idx = tid; idx < n16; idx += nthreads) {
    __builtin_amdgcn_global_load_async_to_lds_b128(
        (gv4i_p)(gsrc + (size_t)idx * 8),
        (lv4i_p)(ldst + (size_t)idx * 8), 0, 0);
  }
#else
  for (int idx = tid; idx < n16; idx += nthreads)
    *(uint4*)(ldst + (size_t)idx * 8) = *(const uint4*)(gsrc + (size_t)idx * 8);
#endif
}

__device__ __forceinline__ void stage_wait() {
#ifdef USE_ASYNC_LDS
  __builtin_amdgcn_s_wait_asynccnt(0);
#endif
  __syncthreads();
}

// ---------------------------------------------------------------------------
// Recurrent scan.  grid = 32 WGs (one 16-column tile of H each), block = 192
// (6 waves: wave = gate(0..2) x batch-half(0..1)).
//   LDS (dynamic, ~86 KB): Whh slice 48 KB (resident all 1024 steps),
//   h bf16 32 KB (async-staged every step), gate tiles 6 KB, lens.
// WMMA inner loop is fully LDS-fed (ds_load_b128 -> v_wmma).  h exchanged
// via L2 with an atomic grid barrier between steps.
// ---------------------------------------------------------------------------
#define SW_ELEMS   (3 * 16 * H_)            // 24576 bf16 = 48 KB
#define SH_ELEMS   (B_ * H_)                // 16384 bf16 = 32 KB
#define SG_ELEMS   (3 * B_ * 16)            // 1536 f32  = 6 KB
#define SCAN_SMEM  (SW_ELEMS * 2 + SH_ELEMS * 2 + SG_ELEMS * 4 + 256)

__global__ __launch_bounds__(192) void scan_kernel(
    const float*  __restrict__ xp,    // [B*T, G]  (includes bih)
    const __bf16* __restrict__ whh,   // [G, H] this layer
    const float*  __restrict__ bhh,   // [G]
    const long long* __restrict__ lens, // [B]
    float*  __restrict__ h32,         // [B, H]
    __bf16* __restrict__ hbf,         // [B, H]
    float*  __restrict__ y,           // [B*T, H]
    float*  __restrict__ finals,      // [B, H] slice of d_out
    unsigned* __restrict__ bar)       // [2] {arrive, phase}
{
  extern __shared__ __align__(16) char smem[];
  __bf16* sw  = (__bf16*)smem;                                  // Whh B-tiles
  __bf16* sh  = (__bf16*)(smem + SW_ELEMS * 2);                 // h (bf16)
  float*  sg  = (float*)(smem + SW_ELEMS * 2 + SH_ELEMS * 2);   // gate tiles
  int*   slen = (int*)(smem + SW_ELEMS * 2 + SH_ELEMS * 2 + SG_ELEMS * 4);

  const int jt   = blockIdx.x;       // column tile 0..31
  const int tid  = threadIdx.x;
  const int lane = tid & 31;
  const int wave = tid >> 5;         // 0..5

  // Stage Whh rows {g*H + jt*16 + n : g<3, n<16} into LDS (async, row-major
  // source is contiguous per row so a strided per-row copy keeps it simple).
  for (int idx = tid; idx < 3 * 16; idx += blockDim.x) {
    const __bf16* src = whh + (size_t)((idx / 16) * H_ + jt * 16 + (idx & 15)) * H_;
    // one row = 512 bf16 = 64 x 16B chunks, issued by this thread
#ifdef USE_ASYNC_LDS
    for (int c = 0; c < H_ / 8; ++c)
      __builtin_amdgcn_global_load_async_to_lds_b128(
          (gv4i_p)(src + c * 8),
          (lv4i_p)(sw + (size_t)idx * H_ + c * 8), 0, 0);
#else
    for (int c = 0; c < H_ / 8; ++c)
      *(uint4*)(sw + (size_t)idx * H_ + c * 8) = *(const uint4*)(src + c * 8);
#endif
  }
  for (int i = tid; i < B_; i += blockDim.x) slen[i] = (int)lens[i];
  stage_wait();

  const int   g     = wave >> 1;      // gate
  const int   mtile = wave & 1;       // batch half
  const int   bn    = lane & 15;
  const int   bkb   = (lane >> 4) * 8;
  const int   arow  = mtile * 16 + (lane & 15);
  const int   akb   = (lane >> 4) * 8;
  const float bhh_n = bhh[g * H_ + jt * 16 + bn];

  unsigned phase = 0;

  for (int t = 0; t < T_; ++t) {
    // ---- async-stage previous h (bf16, 32 KB) into LDS once per step ----
    stage_to_lds(hbf, sh, SH_ELEMS / 8, tid, blockDim.x);
    stage_wait();

    // ---- gh tile: (32 x 16) slice of h @ Whh^T, K = 512, all LDS-fed ----
    v8f acc = {};
    const __bf16* hrow = sh + (size_t)arow * H_;
    #pragma unroll
    for (int k0 = 0; k0 < H_; k0 += 32) {
      union { v16bf v; uint4 q[2]; } a, b;
      a.q[0] = *(const uint4*)(hrow + k0 + akb);
      a.q[1] = *(const uint4*)(hrow + k0 + akb + 16);
      const __bf16* bp = &sw[(g * 16 + bn) * H_ + k0 + bkb];
      b.q[0] = *(const uint4*)(bp);
      b.q[1] = *(const uint4*)(bp + 16);
      acc = __builtin_amdgcn_wmma_f32_16x16x32_bf16(false, a.v, false, b.v,
                                                    (short)0, acc, false, false);
    }
    const int mbase = mtile * 16 + (lane >> 4) * 8;
    #pragma unroll
    for (int r = 0; r < 8; ++r)
      sg[(g * B_ + mbase + r) * 16 + bn] = acc[r] + bhh_n;
    __syncthreads();

    // ---- GRU gates for this WG's 32x16 h slice ----
    for (int idx = tid; idx < B_ * 16; idx += blockDim.x) {
      int b = idx >> 4;
      int n = idx & 15;
      int col = jt * 16 + n;
      size_t rowoff = ((size_t)b * T_ + t) * G_;
      float xr = xp[rowoff + col];
      float xz = xp[rowoff + H_ + col];
      float xn = xp[rowoff + 2 * H_ + col];
      float hr = sg[(0 * B_ + b) * 16 + n];
      float hz = sg[(1 * B_ + b) * 16 + n];
      float hn = sg[(2 * B_ + b) * 16 + n];
      float r  = 1.0f / (1.0f + __expf(-(xr + hr)));
      float z  = 1.0f / (1.0f + __expf(-(xz + hz)));
      float nn = tanhf(xn + r * hn);
      float hold = h32[b * H_ + col];
      float hnew = (1.0f - z) * nn + z * hold;
      bool  m    = t < slen[b];
      float hkeep = m ? hnew : hold;
      y[((size_t)b * T_ + t) * H_ + col] = m ? hnew : 0.0f;
      h32[b * H_ + col] = hkeep;
      hbf[b * H_ + col] = (__bf16)hkeep;
      if (t + 1 < T_)  // stream next step's xp rows toward the WGP
        __builtin_prefetch(&xp[rowoff + G_ + col], 0, 1);
    }

    // ---- device-wide step barrier across the 32 resident WGs ----
    __syncthreads();
    if (tid == 0) {
      __threadfence();
      if (atomicAdd(&bar[0], 1u) == gridDim.x - 1) {
        bar[0] = 0u;
        __threadfence();
        atomicAdd(&bar[1], 1u);
      } else {
        while (atomicAdd(&bar[1], 0u) <= phase) { __builtin_amdgcn_s_sleep(0); }
      }
      phase++;
    }
    __syncthreads();
  }

  // final hidden state for this column tile
  for (int idx = tid; idx < B_ * 16; idx += blockDim.x) {
    int b = idx >> 4;
    int col = jt * 16 + (idx & 15);
    finals[b * H_ + col] = h32[b * H_ + col];
  }
}

// ---------------------------------------------------------------------------
// LayerNorm over H=512 per (b,t) row.  Writes bf16 for the next layer's GEMM
// and/or fp32 to d_out for the final layer.
// ---------------------------------------------------------------------------
__global__ __launch_bounds__(256) void ln_kernel(
    const float* __restrict__ yin,    // [B*T, H]
    const float* __restrict__ gamma,  // [H]
    const float* __restrict__ beta,   // [H]
    __bf16* __restrict__ next_bf,     // [B*T, H] or nullptr
    float*  __restrict__ out_f32)     // [B*T, H] or nullptr
{
  __shared__ float red[256];
  const int row = blockIdx.x;
  const int tid = threadIdx.x;
  const float* yr = yin + (size_t)row * H_;
  float v0 = yr[tid];
  float v1 = yr[tid + 256];

  red[tid] = v0 + v1;
  __syncthreads();
  for (int s = 128; s > 0; s >>= 1) {
    if (tid < s) red[tid] += red[tid + s];
    __syncthreads();
  }
  float mu = red[0] * (1.0f / H_);
  __syncthreads();

  float d0 = v0 - mu, d1 = v1 - mu;
  red[tid] = d0 * d0 + d1 * d1;
  __syncthreads();
  for (int s = 128; s > 0; s >>= 1) {
    if (tid < s) red[tid] += red[tid + s];
    __syncthreads();
  }
  float rstd = rsqrtf(red[0] * (1.0f / H_) + LN_EPS);

  float o0 = d0 * rstd * gamma[tid] + beta[tid];
  float o1 = d1 * rstd * gamma[tid + 256] + beta[tid + 256];
  if (out_f32) {
    out_f32[(size_t)row * H_ + tid]       = o0;
    out_f32[(size_t)row * H_ + tid + 256] = o1;
  }
  if (next_bf) {
    next_bf[(size_t)row * H_ + tid]       = (__bf16)o0;
    next_bf[(size_t)row * H_ + tid + 256] = (__bf16)o1;
  }
}

// ---------------------------------------------------------------------------
// Host orchestration (graph-capture safe: only kernel launches on `stream`)
// Workspace layout (~312 MB):
//   xin_bf [B*T*H bf16] | xp [B*T*G f32] | wih_bf | whh_bf [L*G*H bf16 each]
//   | h32 [B*H f32] | hbf [B*H bf16] | ybuf [B*T*H f32] | barrier words
// ---------------------------------------------------------------------------
extern "C" void kernel_launch(void* const* d_in, const int* in_sizes, int n_in,
                              void* d_out, int out_size, void* d_ws, size_t ws_size,
                              hipStream_t stream) {
  (void)in_sizes; (void)n_in; (void)out_size; (void)ws_size;
  const float*     x      = (const float*)d_in[0];
  const long long* x_lens = (const long long*)d_in[1];
  const float*     h0     = (const float*)d_in[2];
  const float*     Wih    = (const float*)d_in[3];
  const float*     Whh    = (const float*)d_in[4];
  const float*     bih    = (const float*)d_in[5];
  const float*     bhh    = (const float*)d_in[6];
  const float*     gamma  = (const float*)d_in[7];
  const float*     beta   = (const float*)d_in[8];

  float* out    = (float*)d_out;
  float* finals = out + (size_t)B_ * T_ * H_;

  char* ws = (char*)d_ws;
  size_t off = 0;
  __bf16* xin_bf = (__bf16*)(ws + off); off += (size_t)B_ * T_ * H_ * 2;
  float*  xp     = (float*) (ws + off); off += (size_t)B_ * T_ * G_ * 4;
  __bf16* wih_bf = (__bf16*)(ws + off); off += (size_t)L_ * G_ * H_ * 2;
  __bf16* whh_bf = (__bf16*)(ws + off); off += (size_t)L_ * G_ * H_ * 2;
  float*  h32    = (float*) (ws + off); off += (size_t)B_ * H_ * 4;
  __bf16* hbf    = (__bf16*)(ws + off); off += (size_t)B_ * H_ * 2 + 256;
  float*  ybuf   = (float*) (ws + off); off += (size_t)B_ * T_ * H_ * 4;
  unsigned* bar  = (unsigned*)(ws + off);

  const int nW = L_ * G_ * H_;
  cast_f32_to_bf16_kernel<<<512,  256, 0, stream>>>(Wih, wih_bf, nW);
  cast_f32_to_bf16_kernel<<<512,  256, 0, stream>>>(Whh, whh_bf, nW);
  cast_f32_to_bf16_kernel<<<1024, 256, 0, stream>>>(x, xin_bf, B_ * T_ * H_);

  for (int l = 0; l < L_; ++l) {
    xproj_kernel<<<(2048 * 96) / 8, 256, 0, stream>>>(
        xin_bf, wih_bf + (size_t)l * G_ * H_, bih + (size_t)l * G_, xp);
    init_layer_kernel<<<(B_ * H_ + 255) / 256, 256, 0, stream>>>(
        h0 + (size_t)l * B_ * H_, h32, hbf, bar);
    scan_kernel<<<32, 192, SCAN_SMEM, stream>>>(
        xp, whh_bf + (size_t)l * G_ * H_, bhh + (size_t)l * G_, x_lens,
        h32, hbf, ybuf, finals + (size_t)l * B_ * H_, bar);
    ln_kernel<<<B_ * T_, 256, 0, stream>>>(
        ybuf, gamma + (size_t)l * H_, beta + (size_t)l * H_,
        (l < L_ - 1) ? xin_bf : (__bf16*)nullptr,
        (l == L_ - 1) ? out : (float*)nullptr);
  }
}
